// EdgePNARegressor_66013647339605
// MI455X (gfx1250) — matrix operand order repaired and checked by reading, not compiled, not run on hardware
//
#include <hip/hip_runtime.h>
#include <hip/hip_bf16.h>
#include <stdint.h>
#include <stddef.h>

typedef __attribute__((ext_vector_type(16))) __bf16 v16bf;
typedef __attribute__((ext_vector_type(8)))  __bf16 v8bf;
typedef __attribute__((ext_vector_type(8)))  float  v8f;

// ---------------------------------------------------------------------------
// CDNA5 hardware f32 atomics (no-return, STOREcnt-tracked) via inline asm.
// ---------------------------------------------------------------------------
__device__ __forceinline__ void atomic_add_f32(float* p, float v) {
  asm volatile("global_atomic_add_f32 %0, %1, off"
               :: "v"((unsigned long long)(uintptr_t)p), "v"(v) : "memory");
}
__device__ __forceinline__ void atomic_min_f32(float* p, float v) {
  asm volatile("global_atomic_min_num_f32 %0, %1, off"
               :: "v"((unsigned long long)(uintptr_t)p), "v"(v) : "memory");
}
__device__ __forceinline__ void atomic_max_f32(float* p, float v) {
  asm volatile("global_atomic_max_num_f32 %0, %1, off"
               :: "v"((unsigned long long)(uintptr_t)p), "v"(v) : "memory");
}

// Assemble a 16-wide bf16 WMMA operand from two aligned 8-element runs.
__device__ __forceinline__ v16bf frag_from_runs(const __bf16* lo, const __bf16* hi) {
  v8bf a = *(const v8bf*)lo;
  v8bf b = *(const v8bf*)hi;
  return __builtin_shufflevector(a, b, 0, 1, 2, 3, 4, 5, 6, 7,
                                 8, 9, 10, 11, 12, 13, 14, 15);
}

// ---------------------------------------------------------------------------
// Utility kernels
// ---------------------------------------------------------------------------
__global__ void fill_kernel(float* __restrict__ p, float val, int n) {
  int i = blockIdx.x * blockDim.x + threadIdx.x;
  if (i < n) p[i] = val;
}

__global__ void degree_kernel(const int* __restrict__ tgt, float* __restrict__ deg, int nE) {
  int e = blockIdx.x * blockDim.x + threadIdx.x;
  if (e < nE) atomic_add_f32(deg + tgt[e], 1.0f);
}

// Generic transpose + f32->bf16: in [Tn][K][F] -> out [Tn*F][K] (column-major
// per tower). Used for pre_W (Tn=4, K=2*Fin, F=Ftower), post_W (Tn=4), lin_W
// (Tn=1). Output row n = t*F + f holds K contiguous bf16 -> B-fragments become
// two global_load_b128 per WMMA.
__global__ void conv_t_bf16(const float* __restrict__ in, __bf16* __restrict__ out,
                            int Tn, int K, int F) {
  int i = blockIdx.x * blockDim.x + threadIdx.x;
  int total = Tn * K * F;
  if (i < total) {
    int t = i / (K * F);
    int r = i % (K * F);
    int k = r / F;
    int f = r % F;
    out[(size_t)(t * F + f) * K + k] = (__bf16)in[i];
  }
}

// ---------------------------------------------------------------------------
// Edge message GEMM + scatter aggregation.
//   A: 16 edges x K2 (=2*XDIM) of concat(x[tgt], x[src])   (bf16 in LDS)
//   B: W' [FOUT][K2] bf16 (column-major)
//   Output elements scattered with sum/sumsq/min/max atomics per target node.
// ---------------------------------------------------------------------------
template<int XDIM, int FOUT, int WAVES>
__global__ __launch_bounds__(WAVES * 32)
void msg_kernel(const float* __restrict__ x,
                const int* __restrict__ srcI, const int* __restrict__ tgtI,
                const __bf16* __restrict__ Wm, const float* __restrict__ bias,
                float* __restrict__ asum, float* __restrict__ asq,
                float* __restrict__ amn, float* __restrict__ amx, int nE) {
  constexpr int K2 = 2 * XDIM;
  __shared__ __bf16 tile[WAVES][16 * K2];

  const int tid  = threadIdx.x;
  const int wave = tid >> 5;
  const int lane = tid & 31;
  const int ebase = (blockIdx.x * WAVES + wave) * 16;

  // ---- stage 16-edge tile: row i = edge, cols [0,XDIM)=x[tgt], [XDIM,2XDIM)=x[src]
  {
    const int i    = lane >> 1;   // edge within group (0..15)
    const int part = lane & 1;    // 0: target row, 1: source row
    const int e    = ebase + i;
    const int node = part ? srcI[e] : tgtI[e];
    const float* xr = x + (size_t)node * XDIM;
    __bf16* dst = &tile[wave][i * K2 + part * XDIM];
    for (int c = 0; c < XDIM; c += 4) {
      float4 f4 = *(const float4*)(xr + c);
      dst[c + 0] = (__bf16)f4.x; dst[c + 1] = (__bf16)f4.y;
      dst[c + 2] = (__bf16)f4.z; dst[c + 3] = (__bf16)f4.w;
    }
  }
  __syncthreads();

  const int half = lane >> 4;
  const int rl   = lane & 15;
  const __bf16* arow = &tile[wave][rl * K2];

  for (int nt = 0; nt < FOUT / 16; ++nt) {
    const __bf16* brow = Wm + (size_t)(nt * 16 + rl) * K2;
    v8f acc = {0.f, 0.f, 0.f, 0.f, 0.f, 0.f, 0.f, 0.f};
    for (int kk = 0; kk < K2 / 32; ++kk) {
      const int ka0 = kk * 32 + 8 * half;    // A: runs [ka0, ka0+8) and [ka0+16, ka0+24)
      const int kb0 = kk * 32 + 8 * half;    // B: runs [kb0, kb0+8) and [kb0+16, kb0+24)
      v16bf va = frag_from_runs(arow + ka0, arow + ka0 + 16);
      v16bf vb = frag_from_runs(brow + kb0, brow + kb0 + 16);
      acc = __builtin_amdgcn_wmma_f32_16x16x32_bf16(false, va, false, vb,
                                                    (short)0, acc, false, false);
    }
    // C layout: VGPR r, lane L -> row = r + 8*(L>>4), col = L&15
    const int col = nt * 16 + rl;
    const float bv = bias[col];
#pragma unroll
    for (int r = 0; r < 8; ++r) {
      const int m    = r + 8 * half;
      const int node = tgtI[ebase + m];
      const float mv = acc[r] + bv;
      const size_t o = (size_t)node * FOUT + col;
      atomic_add_f32(asum + o, mv);
      atomic_add_f32(asq  + o, mv * mv);
      atomic_min_f32(amn  + o, mv);
      atomic_max_f32(amx  + o, mv);
    }
  }
}

// ---------------------------------------------------------------------------
// Post-aggregation per-tower GEMM.
// Full post_in rows [T][16][KP] are materialized in dynamic LDS at stage time
// (x | agg | agg*amp | agg*att, agg = mean,min,max,std), so every A-fragment
// is two ds_load_b128 and the scaler math runs once per element.
// ---------------------------------------------------------------------------
template<int XDIM, int F>
__global__ __launch_bounds__(256)
void post_kernel(const float* __restrict__ xin, const float* __restrict__ deg,
                 const float* __restrict__ asum, const float* __restrict__ asq,
                 const float* __restrict__ amn, const float* __restrict__ amx,
                 const __bf16* __restrict__ Wp, const float* __restrict__ bias,
                 const float* __restrict__ avg_log_p,
                 float* __restrict__ hmid, int nN) {
  constexpr int T  = 4;
  constexpr int A4 = 4 * F;            // mean|min|max|std block per tower
  constexpr int KP = XDIM + 12 * F;    // per-tower post input dim (mult of 32)

  extern __shared__ char smem[];
  __bf16* pin = (__bf16*)smem;         // [T][16][KP]
  __shared__ float degcS[16], hasS[16], ampS[16], attS[16];

  const int tid  = threadIdx.x;
  const int tile = blockIdx.x;
  auto nodeOf = [&](int row) { int n = tile * 16 + row; return n < nN ? n : nN - 1; };

  if (tid < 16) {
    const int n = nodeOf(tid);
    const float d = deg[n];
    const float dlog = logf(fmaxf(d, 1.0f) + 1.0f);
    const float al = avg_log_p[0];
    degcS[tid] = fmaxf(d, 1.0f);
    hasS[tid]  = (d > 0.f) ? 1.f : 0.f;
    ampS[tid]  = dlog / al;
    attS[tid]  = al / dlog;
  }
  __syncthreads();

  for (int idx = tid; idx < T * 16 * KP; idx += 256) {
    const int t   = idx / (16 * KP);
    const int r   = idx % (16 * KP);
    const int row = r / KP;
    const int k   = r % KP;
    const int n   = nodeOf(row);
    float val;
    if (k < XDIM) {
      val = xin[(size_t)n * XDIM + k];
    } else {
      const int q = k - XDIM;
      const int s = q / A4;            // 0:identity 1:amplification 2:attenuation
      const int rr = q % A4;
      const int a = rr / F;            // 0:mean 1:min 2:max 3:std
      const int f = rr % F;
      const size_t o = (size_t)n * (T * F) + t * F + f;
      const float degc = degcS[row];
      if (a == 0) {
        val = asum[o] / degc;
      } else if (a == 1) {
        val = amn[o] * hasS[row];
      } else if (a == 2) {
        val = amx[o] * hasS[row];
      } else {
        const float mean = asum[o] / degc;
        const float var  = fmaxf(asq[o] / degc - mean * mean, 0.f);
        val = sqrtf(var + 1e-5f);
      }
      if (s == 1)      val *= ampS[row];
      else if (s == 2) val *= attS[row];
    }
    pin[idx] = (__bf16)val;
  }
  __syncthreads();

  const int wave = tid >> 5, lane = tid & 31;
  const int half = lane >> 4, rl = lane & 15;
  constexpr int NT = F / 16;

  for (int task = wave; task < T * NT; task += 8) {
    const int t  = task / NT;
    const int nt = task % NT;
    const __bf16* arow = pin + (size_t)(t * 16 + rl) * KP;
    const __bf16* brow = Wp + (size_t)(t * F + nt * 16 + rl) * KP;
    v8f acc = {0.f, 0.f, 0.f, 0.f, 0.f, 0.f, 0.f, 0.f};
    for (int kk = 0; kk < KP / 32; ++kk) {
      const int ka0 = kk * 32 + 8 * half;
      const int kb0 = kk * 32 + 8 * half;
      v16bf va = frag_from_runs(arow + ka0, arow + ka0 + 16);
      v16bf vb = frag_from_runs(brow + kb0, brow + kb0 + 16);
      acc = __builtin_amdgcn_wmma_f32_16x16x32_bf16(false, va, false, vb,
                                                    (short)0, acc, false, false);
    }
    const int colL = nt * 16 + rl;
    const float bv = bias[t * F + colL];
#pragma unroll
    for (int r8 = 0; r8 < 8; ++r8) {
      const int row = r8 + 8 * half;
      const int n = tile * 16 + row;
      if (n < nN) hmid[(size_t)n * (T * F) + t * F + colL] = acc[r8] + bv;
    }
  }
}

// ---------------------------------------------------------------------------
// Dense lin GEMM [N x K] @ W'[O][K] + bias, nan_to_num, BN sum/sumsq atomics.
// ---------------------------------------------------------------------------
template<int K, int O>
__global__ __launch_bounds__(256)
void lin_kernel(const float* __restrict__ hin, const __bf16* __restrict__ Wl,
                const float* __restrict__ bias, float* __restrict__ hout,
                float* __restrict__ bnsum, float* __restrict__ bnsq, int nN) {
  __shared__ __bf16 xs[16 * K];
  const int tid  = threadIdx.x;
  const int tile = blockIdx.x;
  auto nodeOf = [&](int row) { int n = tile * 16 + row; return n < nN ? n : nN - 1; };

  for (int idx = tid; idx < 16 * K; idx += 256) {
    const int row = idx / K, c = idx % K;
    xs[idx] = (__bf16)hin[(size_t)nodeOf(row) * K + c];
  }
  __syncthreads();

  const int wave = tid >> 5, lane = tid & 31;
  const int half = lane >> 4, rl = lane & 15;
  const __bf16* arow = &xs[rl * K];

  for (int nt = wave; nt < O / 16; nt += 8) {
    const __bf16* brow = Wl + (size_t)(nt * 16 + rl) * K;
    v8f acc = {0.f, 0.f, 0.f, 0.f, 0.f, 0.f, 0.f, 0.f};
    for (int kk = 0; kk < K / 32; ++kk) {
      const int ka0 = kk * 32 + 8 * half;
      const int kb0 = kk * 32 + 8 * half;
      v16bf va = frag_from_runs(arow + ka0, arow + ka0 + 16);
      v16bf vb = frag_from_runs(brow + kb0, brow + kb0 + 16);
      acc = __builtin_amdgcn_wmma_f32_16x16x32_bf16(false, va, false, vb,
                                                    (short)0, acc, false, false);
    }
    const int col = nt * 16 + rl;
    const float bv = bias[col];
#pragma unroll
    for (int r8 = 0; r8 < 8; ++r8) {
      const int row = r8 + 8 * half;
      const int n = tile * 16 + row;
      if (n < nN) {
        float v = acc[r8] + bv;
        if (!(v == v)) v = 0.f;                           // nan_to_num
        v = fminf(fmaxf(v, -3.0e38f), 3.0e38f);
        hout[(size_t)n * O + col] = v;
        atomic_add_f32(bnsum + col, v);
        atomic_add_f32(bnsq + col, v * v);
      }
    }
  }
}

// ---------------------------------------------------------------------------
// BatchNorm (training stats from sums) + ReLU
// ---------------------------------------------------------------------------
__global__ void bn_relu_kernel(const float* __restrict__ hin,
                               const float* __restrict__ bnsum, const float* __restrict__ bnsq,
                               const float* __restrict__ g, const float* __restrict__ b,
                               float* __restrict__ out, int nN, int C) {
  int i = blockIdx.x * blockDim.x + threadIdx.x;
  if (i < nN * C) {
    const int c = i % C;
    const float inv = 1.0f / (float)nN;
    const float m = bnsum[c] * inv;
    const float var = fmaxf(bnsq[c] * inv - m * m, 0.f);
    const float v = g[c] * (hin[i] - m) * rsqrtf(var + 1e-5f) + b[c];
    out[i] = fmaxf(v, 0.f);
  }
}

// ---------------------------------------------------------------------------
// Host orchestration
// ---------------------------------------------------------------------------
extern "C" void kernel_launch(void* const* d_in, const int* in_sizes, int n_in,
                              void* d_out, int out_size, void* d_ws, size_t ws_size,
                              hipStream_t stream) {
  const int N = 25000, E = 400000;
  const int IN = 128, HID = 256, OUT = 128, T = 4, F1 = 64, F2 = 32;
  const int KP1 = IN + 12 * F1;    // 896
  const int KP2 = HID + 12 * F2;   // 640

  const float* x       = (const float*)d_in[0];
  const int*   ei      = (const int*)d_in[1];
  const int*   srcI    = ei;
  const int*   tgtI    = ei + E;
  const float* avg_log = (const float*)d_in[2];
  const float* pre1_W  = (const float*)d_in[3];
  const float* pre1_b  = (const float*)d_in[4];
  const float* post1_W = (const float*)d_in[5];
  const float* post1_b = (const float*)d_in[6];
  const float* lin1_W  = (const float*)d_in[7];
  const float* lin1_b  = (const float*)d_in[8];
  const float* bn1_g   = (const float*)d_in[9];
  const float* bn1_b   = (const float*)d_in[10];
  const float* pre2_W  = (const float*)d_in[11];
  const float* pre2_b  = (const float*)d_in[12];
  const float* post2_W = (const float*)d_in[13];
  const float* post2_b = (const float*)d_in[14];
  const float* lin2_W  = (const float*)d_in[15];
  const float* lin2_b  = (const float*)d_in[16];
  const float* bn2_g   = (const float*)d_in[17];
  const float* bn2_b   = (const float*)d_in[18];

  // ---- workspace carve-up (256B aligned regions)
  char* ws = (char*)d_ws;
  size_t off = 0;
  auto take = [&](size_t bytes) -> char* {
    char* p = ws + off;
    off += (bytes + 255) & ~(size_t)255;
    return p;
  };
  float* deg   = (float*)take((size_t)N * 4);
  float* asum  = (float*)take((size_t)N * HID * 4);
  float* asq   = (float*)take((size_t)N * HID * 4);
  float* amn   = (float*)take((size_t)N * HID * 4);
  float* amx   = (float*)take((size_t)N * HID * 4);
  float* hmid  = (float*)take((size_t)N * HID * 4);
  float* hlin  = (float*)take((size_t)N * HID * 4);
  float* hact  = (float*)take((size_t)N * HID * 4);
  float* bnsum = (float*)take((size_t)HID * 4);
  float* bnsq  = (float*)take((size_t)HID * 4);
  __bf16* W1m  = (__bf16*)take((size_t)2 * IN * (T * F1) * 2);   // [256 n][256 k]
  __bf16* Wp1m = (__bf16*)take((size_t)T * KP1 * F1 * 2);        // [4*64 n][896 k]
  __bf16* L1m  = (__bf16*)take((size_t)(T * F1) * HID * 2);      // [256 n][256 k]
  __bf16* W2m  = (__bf16*)take((size_t)2 * HID * (T * F2) * 2);  // [128 n][512 k]
  __bf16* Wp2m = (__bf16*)take((size_t)T * KP2 * F2 * 2);        // [4*32 n][640 k]
  __bf16* L2m  = (__bf16*)take((size_t)(T * F2) * OUT * 2);      // [128 n][128 k]
  (void)ws_size; (void)n_in; (void)in_sizes; (void)out_size;

  auto cdiv = [](int a, int b) { return (a + b - 1) / b; };
  const float BIG = 3.0e38f;

  // ---- init + weight repack
  fill_kernel<<<cdiv(N, 256), 256, 0, stream>>>(deg, 0.f, N);
  fill_kernel<<<cdiv(N * HID, 256), 256, 0, stream>>>(asum, 0.f, N * HID);
  fill_kernel<<<cdiv(N * HID, 256), 256, 0, stream>>>(asq, 0.f, N * HID);
  fill_kernel<<<cdiv(N * HID, 256), 256, 0, stream>>>(amn, BIG, N * HID);
  fill_kernel<<<cdiv(N * HID, 256), 256, 0, stream>>>(amx, -BIG, N * HID);
  fill_kernel<<<1, 256, 0, stream>>>(bnsum, 0.f, HID);
  fill_kernel<<<1, 256, 0, stream>>>(bnsq, 0.f, HID);

  conv_t_bf16<<<cdiv(T * 2 * IN * F1, 256), 256, 0, stream>>>(pre1_W, W1m, T, 2 * IN, F1);
  conv_t_bf16<<<cdiv(T * KP1 * F1, 256), 256, 0, stream>>>(post1_W, Wp1m, T, KP1, F1);
  conv_t_bf16<<<cdiv(T * F1 * HID, 256), 256, 0, stream>>>(lin1_W, L1m, 1, T * F1, HID);
  conv_t_bf16<<<cdiv(T * 2 * HID * F2, 256), 256, 0, stream>>>(pre2_W, W2m, T, 2 * HID, F2);
  conv_t_bf16<<<cdiv(T * KP2 * F2, 256), 256, 0, stream>>>(post2_W, Wp2m, T, KP2, F2);
  conv_t_bf16<<<cdiv(T * F2 * OUT, 256), 256, 0, stream>>>(lin2_W, L2m, 1, T * F2, OUT);

  degree_kernel<<<cdiv(E, 256), 256, 0, stream>>>(tgtI, deg, E);

  const size_t shm1 = (size_t)T * 16 * KP1 * sizeof(__bf16);   // 114688 B
  const size_t shm2 = (size_t)T * 16 * KP2 * sizeof(__bf16);   //  81920 B

  // ================= Layer 1 =================
  msg_kernel<128, 256, 4><<<E / (16 * 4), 128, 0, stream>>>(
      x, srcI, tgtI, W1m, pre1_b, asum, asq, amn, amx, E);
  post_kernel<128, 64><<<cdiv(N, 16), 256, shm1, stream>>>(
      x, deg, asum, asq, amn, amx, Wp1m, post1_b, avg_log, hmid, N);
  lin_kernel<256, 256><<<cdiv(N, 16), 256, 0, stream>>>(
      hmid, L1m, lin1_b, hlin, bnsum, bnsq, N);
  bn_relu_kernel<<<cdiv(N * HID, 256), 256, 0, stream>>>(
      hlin, bnsum, bnsq, bn1_g, bn1_b, hact, N, HID);

  // ---- re-init aggregation state for layer 2 (uses first N*OUT of each buffer)
  fill_kernel<<<cdiv(N * OUT, 256), 256, 0, stream>>>(asum, 0.f, N * OUT);
  fill_kernel<<<cdiv(N * OUT, 256), 256, 0, stream>>>(asq, 0.f, N * OUT);
  fill_kernel<<<cdiv(N * OUT, 256), 256, 0, stream>>>(amn, BIG, N * OUT);
  fill_kernel<<<cdiv(N * OUT, 256), 256, 0, stream>>>(amx, -BIG, N * OUT);
  fill_kernel<<<1, 256, 0, stream>>>(bnsum, 0.f, OUT);
  fill_kernel<<<1, 256, 0, stream>>>(bnsq, 0.f, OUT);

  // ================= Layer 2 =================
  msg_kernel<256, 128, 2><<<E / (16 * 2), 64, 0, stream>>>(
      hact, srcI, tgtI, W2m, pre2_b, asum, asq, amn, amx, E);
  post_kernel<256, 32><<<cdiv(N, 16), 256, shm2, stream>>>(
      hact, deg, asum, asq, amn, amx, Wp2m, post2_b, avg_log, hmid, N);
  lin_kernel<128, 128><<<cdiv(N, 16), 256, 0, stream>>>(
      hmid, L2m, lin2_b, hlin, bnsum, bnsq, N);
  bn_relu_kernel<<<cdiv(N * OUT, 256), 256, 0, stream>>>(
      hlin, bnsum, bnsq, bn2_g, bn2_b, (float*)d_out, N, OUT);
}
